// SelfAttention_7292854468970
// MI455X (gfx1250) — matrix operand order repaired and checked by reading, not compiled, not run on hardware
//
#include <hip/hip_runtime.h>
#include <hip/hip_bf16.h>

typedef __bf16 bf16_t;
typedef __attribute__((ext_vector_type(16))) __bf16 v16bf;
typedef __attribute__((ext_vector_type(8)))  __bf16 v8bf;
typedef __attribute__((ext_vector_type(8)))  float  v8f;

#define BB 4
#define SS 2048
#define DD 1024

// ---- workspace layout (bytes) -------------------------------------------
// xb : [B*S, D]  bf16   (x converted)
// wt : 3 x [D, D] bf16  (row n = column n of W, i.e. W transposed)
// Qb : [B*S, D]  bf16
// Kb : [B*S, D]  bf16
// Vt : [B, D, S] bf16   (V transposed per batch)
// sc : [B, S, S] f32    (scores; softmax overwrites each row in place with
//                        bf16 probs at row pitch 2*S bf16 elements)
static constexpr size_t OFF_XB = 0;
static constexpr size_t OFF_WT = OFF_XB + (size_t)BB * SS * DD * 2;
static constexpr size_t OFF_QB = OFF_WT + (size_t)3 * DD * DD * 2;
static constexpr size_t OFF_KB = OFF_QB + (size_t)BB * SS * DD * 2;
static constexpr size_t OFF_VT = OFF_KB + (size_t)BB * SS * DD * 2;
static constexpr size_t OFF_SC = OFF_VT + (size_t)BB * SS * DD * 2;

// ---- WMMA fragment loaders ----------------------------------------------
// A fragment: 16x32 bf16 tile at (m0,k0) from row-major [lda] matrix.
// lane l holds row m0+(l&15); elements 0..7  <- K = k0 + 8*(l>>4) + 0..7
//                             elements 8..15 <- K = k0 + 16 + 8*(l>>4) + 0..7
__device__ __forceinline__ v16bf load_afrag(const bf16_t* __restrict__ base,
                                            int lda, int m0, int k0, int lane) {
    const int half = lane >> 4;
    const bf16_t* p = base + (size_t)(m0 + (lane & 15)) * (size_t)lda + k0 + half * 8;
    v8bf lo = *(const v8bf*)(p);
    v8bf hi = *(const v8bf*)(p + 16);
    return __builtin_shufflevector(lo, hi, 0, 1, 2, 3, 4, 5, 6, 7,
                                           8, 9, 10, 11, 12, 13, 14, 15);
}

// B fragment: 32x16 bf16 tile; Bt stores B's columns as rows (row n = col n).
// lane l holds column n0+(l&15); elements 0..7 <- K = k0 + 8*(l>>4) + 0..7, 8..15 <- K = k0 + 16 + 8*(l>>4) + 0..7
__device__ __forceinline__ v16bf load_bfrag(const bf16_t* __restrict__ base,
                                            int ldb, int n0, int k0, int lane) {
    const int half = lane >> 4;
    const bf16_t* p = base + (size_t)(n0 + (lane & 15)) * (size_t)ldb + k0 + half * 8;
    v8bf lo = *(const v8bf*)(p);
    v8bf hi = *(const v8bf*)(p + 16);
    return __builtin_shufflevector(lo, hi, 0, 1, 2, 3, 4, 5, 6, 7,
                                           8, 9, 10, 11, 12, 13, 14, 15);
}

__device__ __forceinline__ v8f wmma_bf16(v16bf a, v16bf b, v8f c) {
    return __builtin_amdgcn_wmma_f32_16x16x32_bf16(false, a, false, b,
                                                   (short)0, c, false, false);
}

// ---- kernel 0: convert x -> bf16, W -> bf16 transposed ------------------
__global__ __launch_bounds__(256) void prep_kernel(
    const float* __restrict__ x, const float* __restrict__ Wq,
    const float* __restrict__ Wk, const float* __restrict__ Wv,
    bf16_t* __restrict__ xb, bf16_t* __restrict__ wt_all) {
    const size_t i = (size_t)blockIdx.x * 256 + threadIdx.x;   // < B*S*D
    xb[i] = (bf16_t)x[i];
    if (i < (size_t)DD * DD) {
        const int k = (int)(i / DD);
        const int n = (int)(i % DD);
        const size_t t = (size_t)n * DD + k;
        wt_all[t] = (bf16_t)Wq[i];
        wt_all[(size_t)DD * DD + t] = (bf16_t)Wk[i];
        wt_all[(size_t)2 * DD * DD + t] = (bf16_t)Wv[i];
    }
}

// ---- kernel 1: QKV projection GEMM (z = 0:Q, 1:K, 2:V-transposed) -------
// C[m,n] = sum_k xb[m,k] * W[k,n] + bias[n];  M=B*S, N=K=D
// wave tile 64x64 (4x4 sub-tiles); block = 8 waves = 2(M) x 4(N) -> 128x256
__global__ __launch_bounds__(256) void qkv_gemm_kernel(
    const bf16_t* __restrict__ xb, const bf16_t* __restrict__ wt_all,
    const float* __restrict__ bq, const float* __restrict__ bk,
    const float* __restrict__ bv, bf16_t* __restrict__ Qb,
    bf16_t* __restrict__ Kb, bf16_t* __restrict__ Vt) {
    const int z = blockIdx.z;
    const bf16_t* Wt = wt_all + (size_t)z * DD * DD;
    const float* bias = (z == 0) ? bq : (z == 1) ? bk : bv;

    const int lane = threadIdx.x & 31;
    const int wave = threadIdx.x >> 5;
    const int m0 = blockIdx.y * 128 + (wave & 1) * 64;
    const int n0 = blockIdx.x * 256 + (wave >> 1) * 64;

    v8f acc[4][4] = {};
    for (int k0 = 0; k0 < DD; k0 += 32) {
        v16bf a[4];
#pragma unroll
        for (int i = 0; i < 4; ++i) a[i] = load_afrag(xb, DD, m0 + 16 * i, k0, lane);
#pragma unroll
        for (int j = 0; j < 4; ++j) {
            v16bf b = load_bfrag(Wt, DD, n0 + 16 * j, k0, lane);
#pragma unroll
            for (int i = 0; i < 4; ++i) acc[i][j] = wmma_bf16(a[i], b, acc[i][j]);
        }
    }

    const int half = lane >> 4;
    const int nl = lane & 15;
#pragma unroll
    for (int j = 0; j < 4; ++j) {
        const int n = n0 + 16 * j + nl;
        const float bb = bias[n];
#pragma unroll
        for (int i = 0; i < 4; ++i) {
#pragma unroll
            for (int r = 0; r < 8; ++r) {
                const int m = m0 + 16 * i + half * 8 + r;
                const float v = acc[i][j][r] + bb;
                if (z == 0) {
                    Qb[(size_t)m * DD + n] = (bf16_t)v;
                } else if (z == 1) {
                    Kb[(size_t)m * DD + n] = (bf16_t)v;
                } else {
                    const int bidx = m >> 11;       // m / S
                    const int s = m & (SS - 1);     // m % S
                    Vt[(size_t)bidx * DD * SS + (size_t)n * SS + s] = (bf16_t)v;
                }
            }
        }
    }
}

// ---- kernel 2: scores = Q . K^T * (1/sqrt(D)) ---------------------------
// per batch: M=N=S, K=D.  B-fragments come straight from K rows (contig).
__global__ __launch_bounds__(256) void scores_gemm_kernel(
    const bf16_t* __restrict__ Qb, const bf16_t* __restrict__ Kb,
    float* __restrict__ scores) {
    const int bz = blockIdx.z;
    const bf16_t* Aq = Qb + (size_t)bz * SS * DD;
    const bf16_t* Bk = Kb + (size_t)bz * SS * DD;
    float* out = scores + (size_t)bz * SS * SS;

    const int lane = threadIdx.x & 31;
    const int wave = threadIdx.x >> 5;
    const int m0 = blockIdx.y * 128 + (wave & 1) * 64;
    const int n0 = blockIdx.x * 256 + (wave >> 1) * 64;

    v8f acc[4][4] = {};
    for (int k0 = 0; k0 < DD; k0 += 32) {
        v16bf a[4];
#pragma unroll
        for (int i = 0; i < 4; ++i) a[i] = load_afrag(Aq, DD, m0 + 16 * i, k0, lane);
#pragma unroll
        for (int j = 0; j < 4; ++j) {
            v16bf b = load_bfrag(Bk, DD, n0 + 16 * j, k0, lane);
#pragma unroll
            for (int i = 0; i < 4; ++i) acc[i][j] = wmma_bf16(a[i], b, acc[i][j]);
        }
    }

    const float scale = 0.03125f;  // 1/sqrt(1024)
    const int half = lane >> 4;
    const int nl = lane & 15;
#pragma unroll
    for (int j = 0; j < 4; ++j) {
        const int n = n0 + 16 * j + nl;
#pragma unroll
        for (int i = 0; i < 4; ++i) {
#pragma unroll
            for (int r = 0; r < 8; ++r) {
                const int m = m0 + 16 * i + half * 8 + r;
                out[(size_t)m * SS + n] = acc[i][j][r] * scale;
            }
        }
    }
}

// ---- kernel 3: row softmax, f32 in -> bf16 probs written in place -------
__global__ __launch_bounds__(256) void softmax_kernel(float* __restrict__ scores) {
    const size_t row = blockIdx.x;                 // 0 .. B*S-1
    float* rp = scores + row * SS;
    const int t = threadIdx.x;

    float v[8];
    float mx = -3.0e38f;
#pragma unroll
    for (int i = 0; i < 8; ++i) {
        v[i] = rp[t + 256 * i];
        mx = fmaxf(mx, v[i]);
    }
#pragma unroll
    for (int off = 16; off >= 1; off >>= 1) mx = fmaxf(mx, __shfl_xor(mx, off, 32));
    __shared__ float redm[8];
    if ((t & 31) == 0) redm[t >> 5] = mx;
    __syncthreads();
    mx = redm[0];
#pragma unroll
    for (int w = 1; w < 8; ++w) mx = fmaxf(mx, redm[w]);

    float sum = 0.0f;
#pragma unroll
    for (int i = 0; i < 8; ++i) {
        v[i] = __expf(v[i] - mx);
        sum += v[i];
    }
#pragma unroll
    for (int off = 16; off >= 1; off >>= 1) sum += __shfl_xor(sum, off, 32);
    __shared__ float reds[8];
    if ((t & 31) == 0) reds[t >> 5] = sum;
    __syncthreads();   // also guarantees all f32 reads done before bf16 writes
    sum = 0.0f;
#pragma unroll
    for (int w = 0; w < 8; ++w) sum += reds[w];
    const float inv = 1.0f / sum;

    bf16_t* pp = (bf16_t*)rp;  // bf16 row occupies the front half of f32 row
#pragma unroll
    for (int i = 0; i < 8; ++i) pp[t + 256 * i] = (bf16_t)(v[i] * inv);
}

// ---- kernel 4: out = P . V ----------------------------------------------
// per batch: M=S, N=D, K=S.  A = bf16 probs (row pitch 2*S elements),
// B-fragments come from Vt rows (feature-major, contiguous over keys).
__global__ __launch_bounds__(256) void out_gemm_kernel(
    const float* __restrict__ scores, const bf16_t* __restrict__ Vt,
    float* __restrict__ out) {
    const int bz = blockIdx.z;
    const bf16_t* Ap = (const bf16_t*)(scores + (size_t)bz * SS * SS);
    const bf16_t* Bv = Vt + (size_t)bz * DD * SS;
    float* op = out + (size_t)bz * SS * DD;

    const int lane = threadIdx.x & 31;
    const int wave = threadIdx.x >> 5;
    const int m0 = blockIdx.y * 128 + (wave & 1) * 64;
    const int n0 = blockIdx.x * 256 + (wave >> 1) * 64;

    v8f acc[4][4] = {};
    for (int k0 = 0; k0 < SS; k0 += 32) {
        v16bf a[4];
#pragma unroll
        for (int i = 0; i < 4; ++i) a[i] = load_afrag(Ap, 2 * SS, m0 + 16 * i, k0, lane);
#pragma unroll
        for (int j = 0; j < 4; ++j) {
            v16bf b = load_bfrag(Bv, SS, n0 + 16 * j, k0, lane);
#pragma unroll
            for (int i = 0; i < 4; ++i) acc[i][j] = wmma_bf16(a[i], b, acc[i][j]);
        }
    }

    const int half = lane >> 4;
    const int nl = lane & 15;
#pragma unroll
    for (int j = 0; j < 4; ++j) {
        const int n = n0 + 16 * j + nl;
#pragma unroll
        for (int i = 0; i < 4; ++i) {
#pragma unroll
            for (int r = 0; r < 8; ++r) {
                const int m = m0 + 16 * i + half * 8 + r;
                op[(size_t)m * DD + n] = acc[i][j][r];
            }
        }
    }
}

// ---- host launcher -------------------------------------------------------
extern "C" void kernel_launch(void* const* d_in, const int* in_sizes, int n_in,
                              void* d_out, int out_size, void* d_ws, size_t ws_size,
                              hipStream_t stream) {
    (void)in_sizes; (void)n_in; (void)out_size; (void)ws_size;
    const float* x  = (const float*)d_in[0];
    const float* Wq = (const float*)d_in[1];
    const float* bq = (const float*)d_in[2];
    const float* Wk = (const float*)d_in[3];
    const float* bk = (const float*)d_in[4];
    const float* Wv = (const float*)d_in[5];
    const float* bv = (const float*)d_in[6];

    char* ws = (char*)d_ws;
    bf16_t* xb = (bf16_t*)(ws + OFF_XB);
    bf16_t* wt = (bf16_t*)(ws + OFF_WT);
    bf16_t* Qb = (bf16_t*)(ws + OFF_QB);
    bf16_t* Kb = (bf16_t*)(ws + OFF_KB);
    bf16_t* Vt = (bf16_t*)(ws + OFF_VT);
    float*  sc = (float* )(ws + OFF_SC);
    float* out = (float*)d_out;

    // 0) convert inputs to bf16 (+ transpose weights)
    prep_kernel<<<(BB * SS * DD) / 256, 256, 0, stream>>>(x, Wq, Wk, Wv, xb, wt);

    // 1) Q/K/V projections: M=B*S=8192, N=1024, K=1024; block tile 128x256
    {
        dim3 grid(DD / 256, (BB * SS) / 128, 3);
        qkv_gemm_kernel<<<grid, 256, 0, stream>>>(xb, wt, bq, bk, bv, Qb, Kb, Vt);
    }
    // 2) scores: per batch M=N=2048, K=1024
    {
        dim3 grid(SS / 256, SS / 128, BB);
        scores_gemm_kernel<<<grid, 256, 0, stream>>>(Qb, Kb, sc);
    }
    // 3) softmax rows (in-place bf16 conversion)
    softmax_kernel<<<BB * SS, 256, 0, stream>>>(sc);

    // 4) out = P.V: per batch M=2048, N=1024, K=2048
    {
        dim3 grid(DD / 256, SS / 128, BB);
        out_gemm_kernel<<<grid, 256, 0, stream>>>(sc, Vt, out);
    }
}
